// Gpt2_55671366090994
// MI455X (gfx1250) — compile-verified, matrix operand order checked by reading, not run on hardware
//
#include <hip/hip_runtime.h>

// ---------------- model constants ----------------
#define LAYERS 12
#define DMODEL 768
#define NHEAD  12
#define FFDIM  3072
#define VOCAB  50257
#define SEQ    1024
#define BATCH  2
#define DEPTH  64              // DMODEL / NHEAD
#define MROWS  (BATCH * SEQ)   // 2048

typedef __attribute__((ext_vector_type(16))) __bf16 v16bf;
typedef __attribute__((ext_vector_type(8)))  float  v8f;

union BfFrag { unsigned int u[8]; v16bf v; };

// truncation f32 -> bf16 (1 VALU per element)
static __device__ __forceinline__ unsigned short bt(float f) {
  return (unsigned short)(__float_as_uint(f) >> 16);
}
// pack two f32 into a dword of 2 bf16 (lo = first)
static __device__ __forceinline__ unsigned int pk2(float lo, float hi) {
  return (__float_as_uint(lo) >> 16) | (__float_as_uint(hi) & 0xFFFF0000u);
}

// ---------------- embedding: h = wte[tok] + wpe[s] ----------------
__global__ void embed_kernel(const int* __restrict__ tokens,
                             const float* __restrict__ wte,
                             const float* __restrict__ wpe,
                             float* __restrict__ h) {
  int idx = blockIdx.x * 256 + threadIdx.x;
  if (idx >= MROWS * DMODEL) return;
  int row = idx / DMODEL, d = idx % DMODEL;
  int s = row % SEQ;
  int tok = tokens[row];
  h[idx] = wte[(size_t)tok * DMODEL + d] + wpe[(size_t)s * DMODEL + d];
}

// ---------------- LayerNorm (one row / block) ----------------
__global__ __launch_bounds__(256)
void ln_kernel(const float* __restrict__ x, const float* __restrict__ g,
               const float* __restrict__ b, float* __restrict__ y) {
  __shared__ float red[256];
  int row = blockIdx.x;
  int t = threadIdx.x;
  const float* xr = x + (size_t)row * DMODEL;
  float s = 0.f;
  for (int i = t; i < DMODEL; i += 256) s += xr[i];
  red[t] = s; __syncthreads();
  for (int off = 128; off > 0; off >>= 1) {
    if (t < off) red[t] += red[t + off];
    __syncthreads();
  }
  float mean = red[0] * (1.f / DMODEL);
  __syncthreads();
  float v = 0.f;
  for (int i = t; i < DMODEL; i += 256) { float d = xr[i] - mean; v += d * d; }
  red[t] = v; __syncthreads();
  for (int off = 128; off > 0; off >>= 1) {
    if (t < off) red[t] += red[t + off];
    __syncthreads();
  }
  float rstd = rsqrtf(red[0] * (1.f / DMODEL) + 1e-6f);
  float* yr = y + (size_t)row * DMODEL;
  for (int i = t; i < DMODEL; i += 256) yr[i] = (xr[i] - mean) * rstd * g[i] + b[i];
}

// ------- bf16 WMMA GEMM: C(MxN) = A(MxK)*B(KxN) [+bias][+resid][relu] -------
// M multiple of 128 (M = 2048 for every call); K multiple of 64.
// Non-transposed B requires N multiple of 64 (true for every such call).
// flags: bit0 = relu, bit1 = B stored transposed (NxK row-major, ldb = K);
//        transposed path handles ragged N (logits, N = 50257).
#define BM 128
#define BN 64
#define BK 64

static __device__ __forceinline__ void load_tile(
    const float* __restrict__ A, const float* __restrict__ Bm,
    float4* areg, float4* breg, int m0g, int n0g, int Ntot, int k0,
    int lda, int ldb, int tr, int tc4, bool btrans) {
  const float* Ap = A + (size_t)m0g * lda + k0;
#pragma unroll
  for (int i = 0; i < 8; ++i)
    areg[i] = *(const float4*)(Ap + (size_t)(i * 16 + tr) * lda + tc4);
  if (!btrans) {
    const float* Bp = Bm + (size_t)k0 * ldb + n0g;
#pragma unroll
    for (int i = 0; i < 4; ++i)
      breg[i] = *(const float4*)(Bp + (size_t)(i * 16 + tr) * ldb + tc4);
  } else {
    const float* Bp = Bm + k0;
#pragma unroll
    for (int i = 0; i < 4; ++i) {
      int gn = n0g + i * 16 + tr;
      if (gn < Ntot) {
        breg[i] = *(const float4*)(Bp + (size_t)gn * ldb + tc4);
      } else {
        breg[i].x = 0.f; breg[i].y = 0.f; breg[i].z = 0.f; breg[i].w = 0.f;
      }
    }
  }
}

static __device__ __forceinline__ void store_tile(
    unsigned short (*as)[BK], unsigned short (*bs)[BK],
    const float4* areg, const float4* breg, int tr, int tc4, bool btrans) {
#pragma unroll
  for (int i = 0; i < 8; ++i) {
    uint2 p; p.x = pk2(areg[i].x, areg[i].y); p.y = pk2(areg[i].z, areg[i].w);
    *(uint2*)&as[i * 16 + tr][tc4] = p;
  }
  if (!btrans) {
#pragma unroll
    for (int i = 0; i < 4; ++i) {           // transpose KxN -> [n][k]
      int k = i * 16 + tr;
      bs[tc4 + 0][k] = bt(breg[i].x);
      bs[tc4 + 1][k] = bt(breg[i].y);
      bs[tc4 + 2][k] = bt(breg[i].z);
      bs[tc4 + 3][k] = bt(breg[i].w);
    }
  } else {
#pragma unroll
    for (int i = 0; i < 4; ++i) {
      uint2 p; p.x = pk2(breg[i].x, breg[i].y); p.y = pk2(breg[i].z, breg[i].w);
      *(uint2*)&bs[i * 16 + tr][tc4] = p;
    }
  }
}

__global__ __launch_bounds__(256)
void gemm_wmma(const float* __restrict__ A, const float* __restrict__ Bm,
               const float* __restrict__ bias, const float* __restrict__ resid,
               float* __restrict__ C,
               int Ntot, int Ktot, int lda, int ldb, int ldc, int flags) {
  __shared__ unsigned short As[2][BM][BK];   // 32 KB (double-buffered)
  __shared__ unsigned short Bs[2][BN][BK];   // 16 KB
  int t = threadIdx.x;
  int lane = t & 31, wid = t >> 5;
  int wmp = wid & 3, wnp = wid >> 2;
  int half = lane >> 4, l15 = lane & 15;
  int m0g = blockIdx.y * BM, n0g = blockIdx.x * BN;
  bool relu = (flags & 1) != 0, btrans = (flags & 2) != 0;
  int tr = t >> 4, tc4 = (t & 15) * 4;
  v8f acc00 = {}, acc01 = {}, acc10 = {}, acc11 = {};

  float4 areg[8], breg[4];
  load_tile(A, Bm, areg, breg, m0g, n0g, Ntot, 0, lda, ldb, tr, tc4, btrans);
  store_tile(As[0], Bs[0], areg, breg, tr, tc4, btrans);
  __syncthreads();

  int buf = 0;
  for (int k0 = 0; k0 < Ktot; k0 += BK, buf ^= 1) {
    bool hasNext = (k0 + BK) < Ktot;
    if (hasNext)                          // global loads in flight behind WMMAs
      load_tile(A, Bm, areg, breg, m0g, n0g, Ntot, k0 + BK, lda, ldb, tr, tc4, btrans);

    int ar0 = wmp * 32 + l15, ar1 = ar0 + 16;
    int bc0 = wnp * 32 + l15, bc1 = bc0 + 16;
#pragma unroll
    for (int ks = 0; ks < 2; ++ks) {
      BfFrag a0, a1, b0, b1;
#pragma unroll
      for (int j = 0; j < 8; ++j) {
        int k = ks * 32 + 2 * j + half * 8 + ((j >= 4) ? 8 : 0);  // ISA layout
        a0.u[j] = *(const unsigned int*)&As[buf][ar0][k];
        a1.u[j] = *(const unsigned int*)&As[buf][ar1][k];
        b0.u[j] = *(const unsigned int*)&Bs[buf][bc0][k];
        b1.u[j] = *(const unsigned int*)&Bs[buf][bc1][k];
      }
      acc00 = __builtin_amdgcn_wmma_f32_16x16x32_bf16(false, a0.v, false, b0.v,
                                                      (short)0, acc00, false, false);
      acc01 = __builtin_amdgcn_wmma_f32_16x16x32_bf16(false, a0.v, false, b1.v,
                                                      (short)0, acc01, false, false);
      acc10 = __builtin_amdgcn_wmma_f32_16x16x32_bf16(false, a1.v, false, b0.v,
                                                      (short)0, acc10, false, false);
      acc11 = __builtin_amdgcn_wmma_f32_16x16x32_bf16(false, a1.v, false, b1.v,
                                                      (short)0, acc11, false, false);
    }
    if (hasNext)
      store_tile(As[buf ^ 1], Bs[buf ^ 1], areg, breg, tr, tc4, btrans);
    __syncthreads();
  }

  // ---- epilogue: m_local = r + 8*half, n_local = l15 ----
  int n0 = n0g + wnp * 32 + l15;
  int n1 = n0 + 16;
  float bias0 = bias ? bias[n0 < Ntot ? n0 : 0] : 0.f;
  float bias1 = bias ? bias[n1 < Ntot ? n1 : 0] : 0.f;
#pragma unroll
  for (int mi = 0; mi < 2; ++mi) {
    const v8f& c0 = mi ? acc10 : acc00;
    const v8f& c1 = mi ? acc11 : acc01;
#pragma unroll
    for (int r = 0; r < 8; ++r) {
      int m = m0g + (wmp * 2 + mi) * 16 + r + half * 8;
      float v0 = c0[r] + bias0, v1 = c1[r] + bias1;
      if (resid) {
        if (n0 < Ntot) v0 += resid[(size_t)m * ldc + n0];
        if (n1 < Ntot) v1 += resid[(size_t)m * ldc + n1];
      }
      if (relu) { v0 = fmaxf(v0, 0.f); v1 = fmaxf(v1, 0.f); }
      if (n0 < Ntot) C[(size_t)m * ldc + n0] = v0;
      if (n1 < Ntot) C[(size_t)m * ldc + n1] = v1;
    }
  }
}

// ---------------- fused flash attention (one head, 64 queries / block) -------
__global__ __launch_bounds__(256)
void attn_wmma(const float* __restrict__ qkv, const int* __restrict__ tokens,
               float* __restrict__ ctx) {
  __shared__ unsigned short Qs[64][64];   // [q][d]   (pre-scaled by 1/8)
  __shared__ unsigned short Ks[64][64];   // [key][d]
  __shared__ unsigned short Vt[64][64];   // [d][key]
  __shared__ float          Ss[64][64];   // raw scores
  __shared__ unsigned short Ps[64][64];   // probs (bf16)
  __shared__ float row_m[64], row_l[64], row_scale[64];
  __shared__ int   padk[64];

  int t = threadIdx.x, lane = t & 31, wid = t >> 5;
  int wm = wid & 3, wnb = (wid >> 2) << 1;
  int half = lane >> 4, l15 = lane & 15;
  int tr = t >> 4, tc4 = (t & 15) * 4;
  int qb = blockIdx.x;
  int bh = blockIdx.y;
  int batch = bh / NHEAD, head = bh % NHEAD;
  int q0 = qb * 64;
  const size_t rs = 3 * DMODEL;
  const float* qbase = qkv + (size_t)(batch * SEQ + q0) * rs + head * DEPTH;

  // load Q tile (scaled by 1/sqrt(64)): loads first, then pack
  {
    float4 qreg[4];
#pragma unroll
    for (int i = 0; i < 4; ++i)
      qreg[i] = *(const float4*)(qbase + (size_t)(i * 16 + tr) * rs + tc4);
#pragma unroll
    for (int i = 0; i < 4; ++i) {
      uint2 p; p.x = pk2(qreg[i].x * 0.125f, qreg[i].y * 0.125f);
      p.y = pk2(qreg[i].z * 0.125f, qreg[i].w * 0.125f);
      *(uint2*)&Qs[i * 16 + tr][tc4] = p;
    }
  }
  if (t < 64) { row_m[t] = -1e30f; row_l[t] = 0.f; }

  v8f o0 = {}; v8f o1 = {};

  for (int kb = 0; kb <= qb; ++kb) {      // causal key-block loop
    int k0g = kb * 64;
    const float* kvb = qkv + (size_t)(batch * SEQ + k0g) * rs + head * DEPTH;
    {
      float4 kreg[4], vreg[4];
#pragma unroll
      for (int i = 0; i < 4; ++i) {       // all loads issue first
        const float* src = kvb + (size_t)(i * 16 + tr) * rs + tc4;
        kreg[i] = *(const float4*)(src + DMODEL);
        vreg[i] = *(const float4*)(src + 2 * DMODEL);
      }
#pragma unroll
      for (int i = 0; i < 4; ++i) {       // then pack + LDS store
        int n = i * 16 + tr;
        uint2 p; p.x = pk2(kreg[i].x, kreg[i].y); p.y = pk2(kreg[i].z, kreg[i].w);
        *(uint2*)&Ks[n][tc4] = p;
        Vt[tc4 + 0][n] = bt(vreg[i].x);
        Vt[tc4 + 1][n] = bt(vreg[i].y);
        Vt[tc4 + 2][n] = bt(vreg[i].z);
        Vt[tc4 + 3][n] = bt(vreg[i].w);
      }
    }
    if (t < 64) padk[t] = (tokens[batch * SEQ + k0g + t] == 0) ? 1 : 0;
    __syncthreads();

    // S = Q * K^T  (DEPTH=64 -> two 16x16x32 steps)
    v8f s0 = {}; v8f s1 = {};
#pragma unroll
    for (int ks = 0; ks < 2; ++ks) {
      BfFrag af, b0, b1;
#pragma unroll
      for (int j = 0; j < 8; ++j) {
        int k = ks * 32 + 2 * j + half * 8 + ((j >= 4) ? 8 : 0);
        af.u[j] = *(const unsigned int*)&Qs[wm * 16 + l15][k];
        b0.u[j] = *(const unsigned int*)&Ks[wnb * 16 + l15][k];
        b1.u[j] = *(const unsigned int*)&Ks[(wnb + 1) * 16 + l15][k];
      }
      s0 = __builtin_amdgcn_wmma_f32_16x16x32_bf16(false, af.v, false, b0.v,
                                                   (short)0, s0, false, false);
      s1 = __builtin_amdgcn_wmma_f32_16x16x32_bf16(false, af.v, false, b1.v,
                                                   (short)0, s1, false, false);
    }
#pragma unroll
    for (int r = 0; r < 8; ++r) {
      int m = wm * 16 + r + half * 8;
      Ss[m][wnb * 16 + l15] = s0[r];
      Ss[m][(wnb + 1) * 16 + l15] = s1[r];
    }
    __syncthreads();

    // online softmax (64 row-owner threads)
    if (t < 64) {
      int qi = q0 + t;
      float mold = row_m[t];
      float mnew = mold;
      for (int n = 0; n < 64; ++n) {
        float val = Ss[t][n];
        if ((k0g + n) > qi || padk[n]) val -= 1e9f;
        mnew = fmaxf(mnew, val);
      }
      float scale = __expf(mold - mnew);
      float sum = 0.f;
      for (int n = 0; n < 64; n += 2) {
        float v0 = Ss[t][n], v1 = Ss[t][n + 1];
        if ((k0g + n)     > qi || padk[n])     v0 -= 1e9f;
        if ((k0g + n + 1) > qi || padk[n + 1]) v1 -= 1e9f;
        float e0 = __expf(v0 - mnew), e1 = __expf(v1 - mnew);
        sum += e0 + e1;
        *(unsigned int*)&Ps[t][n] = pk2(e0, e1);
      }
      row_scale[t] = scale;
      row_l[t] = row_l[t] * scale + sum;
      row_m[t] = mnew;
    }
    __syncthreads();

    // rescale running O, then O += P * V
#pragma unroll
    for (int r = 0; r < 8; ++r) {
      int m = wm * 16 + r + half * 8;
      float sc = row_scale[m];
      o0[r] *= sc; o1[r] *= sc;
    }
#pragma unroll
    for (int ks = 0; ks < 2; ++ks) {
      BfFrag af, b0, b1;
#pragma unroll
      for (int j = 0; j < 8; ++j) {
        int k = ks * 32 + 2 * j + half * 8 + ((j >= 4) ? 8 : 0);
        af.u[j] = *(const unsigned int*)&Ps[wm * 16 + l15][k];
        b0.u[j] = *(const unsigned int*)&Vt[wnb * 16 + l15][k];
        b1.u[j] = *(const unsigned int*)&Vt[(wnb + 1) * 16 + l15][k];
      }
      o0 = __builtin_amdgcn_wmma_f32_16x16x32_bf16(false, af.v, false, b0.v,
                                                   (short)0, o0, false, false);
      o1 = __builtin_amdgcn_wmma_f32_16x16x32_bf16(false, af.v, false, b1.v,
                                                   (short)0, o1, false, false);
    }
    __syncthreads();
  }

  // finalize: ctx = O / l
#pragma unroll
  for (int r = 0; r < 8; ++r) {
    int m = wm * 16 + r + half * 8;
    float l = row_l[m];
    float inv = (l > 0.f) ? 1.f / l : 0.f;
    size_t orow = (size_t)(batch * SEQ + q0 + m) * DMODEL + head * DEPTH;
    ctx[orow + wnb * 16 + l15] = o0[r] * inv;
    ctx[orow + (wnb + 1) * 16 + l15] = o1[r] * inv;
  }
}

// ---------------- driver ----------------
extern "C" void kernel_launch(void* const* d_in, const int* in_sizes, int n_in,
                              void* d_out, int out_size, void* d_ws, size_t ws_size,
                              hipStream_t stream) {
  (void)in_sizes; (void)n_in; (void)out_size; (void)ws_size;
  const int*   tokens = (const int*)d_in[0];
  const float* wte    = (const float*)d_in[1];
  const float* wpe    = (const float*)d_in[2];
  const float* ln1_g  = (const float*)d_in[3];
  const float* ln1_b  = (const float*)d_in[4];
  const float* ln2_g  = (const float*)d_in[5];
  const float* ln2_b  = (const float*)d_in[6];
  const float* attn_w = (const float*)d_in[7];
  const float* attn_b = (const float*)d_in[8];
  const float* proj_w = (const float*)d_in[9];
  const float* proj_b = (const float*)d_in[10];
  const float* fc1_w  = (const float*)d_in[11];
  const float* fc1_b  = (const float*)d_in[12];
  const float* fc2_w  = (const float*)d_in[13];
  const float* fc2_b  = (const float*)d_in[14];
  const float* lnf_g  = (const float*)d_in[15];
  const float* lnf_b  = (const float*)d_in[16];
  float* logits = (float*)d_out;

  float* ws = (float*)d_ws;
  float* h   = ws;                                  // M x D
  float* a   = h   + (size_t)MROWS * DMODEL;        // M x D
  float* qkv = a   + (size_t)MROWS * DMODEL;        // M x 3D
  float* ctx = qkv + (size_t)MROWS * 3 * DMODEL;    // M x D
  float* ff  = ctx + (size_t)MROWS * DMODEL;        // M x F

  dim3 blk(256);
  embed_kernel<<<(MROWS * DMODEL + 255) / 256, blk, 0, stream>>>(tokens, wte, wpe, h);

  dim3 gQKV((3 * DMODEL) / BN, MROWS / BM);
  dim3 gPRJ(DMODEL / BN, MROWS / BM);
  dim3 gFC1(FFDIM / BN, MROWS / BM);
  dim3 gATT(SEQ / 64, BATCH * NHEAD);

  for (int l = 0; l < LAYERS; ++l) {
    ln_kernel<<<MROWS, blk, 0, stream>>>(h, ln1_g + (size_t)l * DMODEL,
                                         ln1_b + (size_t)l * DMODEL, a);
    gemm_wmma<<<gQKV, blk, 0, stream>>>(
        a, attn_w + (size_t)l * DMODEL * 3 * DMODEL, attn_b + (size_t)l * 3 * DMODEL,
        nullptr, qkv, 3 * DMODEL, DMODEL, DMODEL, 3 * DMODEL, 3 * DMODEL, 0);
    attn_wmma<<<gATT, blk, 0, stream>>>(qkv, tokens, ctx);
    gemm_wmma<<<gPRJ, blk, 0, stream>>>(
        ctx, proj_w + (size_t)l * DMODEL * DMODEL, proj_b + (size_t)l * DMODEL,
        /*resid=*/h, /*C=*/h, DMODEL, DMODEL, DMODEL, DMODEL, DMODEL, 0);
    ln_kernel<<<MROWS, blk, 0, stream>>>(h, ln2_g + (size_t)l * DMODEL,
                                         ln2_b + (size_t)l * DMODEL, a);
    gemm_wmma<<<gFC1, blk, 0, stream>>>(
        a, fc1_w + (size_t)l * DMODEL * FFDIM, fc1_b + (size_t)l * FFDIM,
        nullptr, ff, FFDIM, DMODEL, DMODEL, FFDIM, FFDIM, /*relu*/1);
    gemm_wmma<<<gPRJ, blk, 0, stream>>>(
        ff, fc2_w + (size_t)l * FFDIM * DMODEL, fc2_b + (size_t)l * DMODEL,
        /*resid=*/h, /*C=*/h, DMODEL, FFDIM, FFDIM, DMODEL, DMODEL, 0);
  }

  ln_kernel<<<MROWS, blk, 0, stream>>>(h, lnf_g, lnf_b, a);

  dim3 gLOG((VOCAB + BN - 1) / BN, MROWS / BM);
  gemm_wmma<<<gLOG, blk, 0, stream>>>(
      a, wte, nullptr, nullptr, logits,
      VOCAB, DMODEL, DMODEL, /*ldb = D (transposed)*/ DMODEL, VOCAB, /*btrans*/2);
}
